// MambaEncoder_36180804502038
// MI455X (gfx1250) — matrix operand order repaired and checked by reading, not compiled, not run on hardware
//
#include <hip/hip_runtime.h>
#include <hip/hip_bf16.h>
#include <math.h>

typedef __attribute__((ext_vector_type(16))) __bf16 v16bf;
typedef __attribute__((ext_vector_type(8)))  float  v8f;

#define B_      4
#define CIN     4
#define SIG     4096
#define D_MODEL 768
#define PATCH   8
#define LAYERS  4
#define DI      1536
#define NSTATE  16
#define DTR     48
#define KCONV   4
#define LSEQ    512          // SIG / PATCH
#define MROWS   2048         // B_ * LSEQ
#define XDW     (DTR + 2*NSTATE)   // 80

// GEMM tiling: 256 threads = 8 waves in 2(M) x 4(N); each wave computes 64x32
#define BM 128
#define BN 128
#define BK 32
#define LDSPAD 8             // 16B pad keeps rows 16B-aligned, spreads banks

// async copy of one 32B chunk: global -> LDS, no VGPR roundtrip (ASYNCcnt)
__device__ __forceinline__ void async_copy32B(uint32_t ldsAddr, const void* gaddr) {
    asm volatile("global_load_async_to_lds_b128 %0, %1, off"
                 :: "v"(ldsAddr), "v"(gaddr) : "memory");
    asm volatile("global_load_async_to_lds_b128 %0, %1, off offset:16"
                 :: "v"(ldsAddr), "v"(gaddr) : "memory");
}
__device__ __forceinline__ void wait_async0() {
    asm volatile("s_wait_asynccnt 0x0" ::: "memory");
}
__device__ __forceinline__ uint32_t lds_addr_of(const void* p) {
    // generic (flat) pointer to __shared__: low 32 bits = LDS byte offset
    return (uint32_t)(uintptr_t)p;
}

// ---------------------------------------------------------------- converts
__global__ void k_f32_to_bf16(const float* __restrict__ s, __bf16* __restrict__ d, int n) {
    int i = blockIdx.x * blockDim.x + threadIdx.x;
    if (i < n) d[i] = (__bf16)s[i];
}

// ---------------------------------------------------------------- patchify
__global__ void k_patchify(const float* __restrict__ x, const float* __restrict__ pw,
                           float* __restrict__ tokF, __bf16* __restrict__ tokB) {
    int i = blockIdx.x * blockDim.x + threadIdx.x;
    if (i >= MROWS * D_MODEL) return;
    int dcol = i % D_MODEL;
    int row  = i / D_MODEL;
    int b = row / LSEQ, l = row % LSEQ;
    const float* xb = x + (size_t)b * CIN * SIG + (size_t)l * PATCH;
    const float* w  = pw + (size_t)dcol * CIN * PATCH;
    float acc = 0.f;
#pragma unroll
    for (int c = 0; c < CIN; ++c)
#pragma unroll
        for (int p = 0; p < PATCH; ++p)
            acc += xb[c * SIG + p] * w[c * PATCH + p];
    tokF[i] = acc;
    tokB[i] = (__bf16)acc;
}

// ---------------------------------------------------------------- WMMA GEMM
// C[M,N] (+)= A[M,K](bf16,row-major,lda) * W[N,K]^T (bf16,row-major,ldw)
// EDGE=false: M%BM==0, N%BN==0, K%BK==0 -> pure async staging, no guards.
// EDGE=true : guarded chunk loads (K%16==0, rows 16B-aligned still assumed).
template <bool EDGE>
__global__ __launch_bounds__(256)
void k_gemm_bf16(const __bf16* __restrict__ A, int lda,
                 const __bf16* __restrict__ W, int ldw,
                 float* C, int ldc, __bf16* Cb,
                 int M, int N, int K, int addC) {
    __shared__ __bf16 sA[2][BM][BK + LDSPAD];
    __shared__ __bf16 sB[2][BN][BK + LDSPAD];
    const int tid   = threadIdx.x;
    const int lane  = tid & 31;
    const int wid   = tid >> 5;
    const int waveM = wid >> 2;       // 0..1  -> 64 rows each
    const int waveN = wid & 3;        // 0..3  -> 32 cols each
    const int blockM = blockIdx.y * BM;
    const int blockN = blockIdx.x * BN;
    const int hi  = lane >> 4;        // lane half (ISA 16-bit fragment layout)
    const int l16 = lane & 15;

    // staging: one 32B chunk (16 bf16) per thread per tile (128x32 both tiles)
    const int sRow = tid >> 1;        // 0..127
    const int sCol = (tid & 1) * 16;  // 0 or 16

    const __bf16* gA = A + (size_t)(blockM + sRow) * lda + sCol;
    const __bf16* gW = W + (size_t)(blockN + sRow) * ldw + sCol;

    v8f acc[4][2] = {};
    const int nsteps = (K + BK - 1) / BK;

    // ---- prologue: stage tile 0 into buffer 0
    if (!EDGE) {
        async_copy32B(lds_addr_of(&sA[0][sRow][sCol]), gA);
        async_copy32B(lds_addr_of(&sB[0][sRow][sCol]), gW);
        wait_async0();
    } else {
        const bool aFull = (blockM + BM <= M) && (BK <= K);
        const bool bFull = (blockN + BN <= N) && (BK <= K);
        bool anyAsync = false;
        if (aFull) { async_copy32B(lds_addr_of(&sA[0][sRow][sCol]), gA); anyAsync = true; }
        else {
            uint4 a0 = {0,0,0,0}, a1 = {0,0,0,0};
            if (blockM + sRow < M && sCol < K) { const uint4* p = (const uint4*)gA; a0 = p[0]; a1 = p[1]; }
            uint4* da = (uint4*)&sA[0][sRow][sCol]; da[0] = a0; da[1] = a1;
        }
        if (bFull) { async_copy32B(lds_addr_of(&sB[0][sRow][sCol]), gW); anyAsync = true; }
        else {
            uint4 b0 = {0,0,0,0}, b1 = {0,0,0,0};
            if (blockN + sRow < N && sCol < K) { const uint4* p = (const uint4*)gW; b0 = p[0]; b1 = p[1]; }
            uint4* db = (uint4*)&sB[0][sRow][sCol]; db[0] = b0; db[1] = b1;
        }
        if (anyAsync) wait_async0();
    }
    __syncthreads();

    for (int s = 0; s < nsteps; ++s) {
        const int  cur  = s & 1;
        const bool more = (s + 1) < nsteps;
        const int  kbn  = (s + 1) * BK;

        // ---- issue next tile's loads (async engine writes LDS; overlaps WMMA)
        uint4 a0 = {0,0,0,0}, a1 = {0,0,0,0}, b0 = {0,0,0,0}, b1 = {0,0,0,0};
        bool aAsync = false, bAsync = false;
        if (!EDGE) {
            if (more) {
                async_copy32B(lds_addr_of(&sA[cur ^ 1][sRow][sCol]), gA + kbn);
                async_copy32B(lds_addr_of(&sB[cur ^ 1][sRow][sCol]), gW + kbn);
            }
        } else {
            const bool kFull = (kbn + BK <= K);
            aAsync = more && (blockM + BM <= M) && kFull;
            bAsync = more && (blockN + BN <= N) && kFull;
            if (aAsync) {
                async_copy32B(lds_addr_of(&sA[cur ^ 1][sRow][sCol]), gA + kbn);
            } else if (more) {
                if (blockM + sRow < M && kbn + sCol < K) {
                    const uint4* p = (const uint4*)(gA + kbn); a0 = p[0]; a1 = p[1];
                }
            }
            if (bAsync) {
                async_copy32B(lds_addr_of(&sB[cur ^ 1][sRow][sCol]), gW + kbn);
            } else if (more) {
                if (blockN + sRow < N && kbn + sCol < K) {
                    const uint4* p = (const uint4*)(gW + kbn); b0 = p[0]; b1 = p[1];
                }
            }
        }
        if (s + 2 < nsteps) {   // global_prefetch_b8 two steps ahead
            __builtin_prefetch(gA + (s + 2) * BK, 0, 3);
            __builtin_prefetch(gW + (s + 2) * BK, 0, 3);
        }

        // ---- fragment gathers (ISA 7.12.2 16-bit layouts) + 8 WMMA
        v16bf afrag[4], bfrag[2];
#pragma unroll
        for (int t = 0; t < 4; ++t) {
            int m = waveM * 64 + t * 16 + l16;       // A 16x32: lane holds row M
#pragma unroll
            for (int e = 0; e < 16; ++e) {
                int k = e + ((e >= 8) ? 8 : 0) + (hi ? 8 : 0);
                afrag[t][e] = sA[cur][m][k];
            }
        }
#pragma unroll
        for (int t = 0; t < 2; ++t) {
            int n = waveN * 32 + t * 16 + l16;       // B 32x16: lane holds col N
#pragma unroll
            for (int e = 0; e < 16; ++e) {
                int k = e + ((e >= 8) ? 8 : 0) + (hi ? 8 : 0);
                bfrag[t][e] = sB[cur][n][k];
            }
        }
#pragma unroll
        for (int ti = 0; ti < 4; ++ti)
#pragma unroll
            for (int tj = 0; tj < 2; ++tj)
                acc[ti][tj] = __builtin_amdgcn_wmma_f32_16x16x32_bf16(
                    false, afrag[ti], false, bfrag[tj],
                    (short)0, acc[ti][tj], false, false);

        // ---- commit any register-staged chunks; order async LDS writes
        if (!EDGE) {
            if (more) wait_async0();
        } else {
            if (more && !aAsync) {
                uint4* da = (uint4*)&sA[cur ^ 1][sRow][sCol]; da[0] = a0; da[1] = a1;
            }
            if (more && !bAsync) {
                uint4* db = (uint4*)&sB[cur ^ 1][sRow][sCol]; db[0] = b0; db[1] = b1;
            }
            if (aAsync || bAsync) wait_async0();
        }
        __syncthreads();
    }

    // ---- epilogue: D layout = 8 VGPRs, lane col = lane&15, row = r + 8*hi
#pragma unroll
    for (int ti = 0; ti < 4; ++ti) {
#pragma unroll
        for (int tj = 0; tj < 2; ++tj) {
            int gn = blockN + waveN * 32 + tj * 16 + l16;
            if (EDGE && gn >= N) continue;
#pragma unroll
            for (int r = 0; r < 8; ++r) {
                int gm = blockM + waveM * 64 + ti * 16 + r + 8 * hi;
                if (EDGE && gm >= M) continue;
                size_t o = (size_t)gm * ldc + gn;
                float v = acc[ti][tj][r];
                if (addC) v += C[o];
                C[o] = v;
                if (Cb) Cb[o] = (__bf16)v;
            }
        }
    }
}

// ---------------------------------------------------------------- depthwise conv + SiLU
__global__ void k_convsilu(const float* __restrict__ xz, const float* __restrict__ cw,
                           const float* __restrict__ cb,
                           float* __restrict__ xrF, __bf16* __restrict__ xrB) {
    int i = blockIdx.x * blockDim.x + threadIdx.x;
    if (i >= MROWS * DI) return;
    int d = i % DI, row = i / DI;
    int b = row / LSEQ, l = row % LSEQ;
    const float* w = cw + (size_t)d * KCONV;
    float acc = cb[d];
#pragma unroll
    for (int j = 0; j < KCONV; ++j) {
        int ls = l + j - (KCONV - 1);
        if (ls >= 0)
            acc += w[j] * xz[(size_t)(b * LSEQ + ls) * (2 * DI) + d];
    }
    float s = acc / (1.f + expf(-acc));     // SiLU
    xrF[i] = s;
    xrB[i] = (__bf16)s;
}

// ---------------------------------------------------------------- softplus(delta + bias)
__global__ void k_softplus(float* dlt, const float* __restrict__ dtb) {
    int i = blockIdx.x * blockDim.x + threadIdx.x;
    if (i >= MROWS * DI) return;
    int d = i % DI;
    float v = dlt[i] + dtb[d];
    dlt[i] = (v > 20.f) ? v : log1pf(expf(v));
}

// ---------------------------------------------------------------- selective scan + gate
// one wave = two channels; lanes (half,n) : n in 0..15 = state index.
__global__ __launch_bounds__(256)
void k_scan(float* dlt,                       // in: delta, out: gated y (f32)
            const float* __restrict__ xd,     // x_dbl (MROWS x 80): B at +48, C at +64
            const float* __restrict__ xr,     // conv output (MROWS x DI)
            const float* __restrict__ xz,     // z = xz[:, DI:2*DI]
            const float* __restrict__ Alog,   // (DI, N) this layer
            const float* __restrict__ Dres,   // (DI)
            __bf16* __restrict__ yB) {
    int wave = blockIdx.x * (blockDim.x >> 5) + (threadIdx.x >> 5);
    int lane = threadIdx.x & 31;
    int half = lane >> 4, n = lane & 15;
    int b = wave / (DI / 2);
    int d = (wave % (DI / 2)) * 2 + half;
    if (b >= B_) return;

    const float A  = -expf(Alog[(size_t)d * NSTATE + n]);
    const float Dr = Dres[d];
    float h = 0.f;
    for (int l = 0; l < LSEQ; ++l) {
        size_t row = (size_t)(b * LSEQ + l);
        float dt = dlt[row * DI + d];
        float x  = xr[row * DI + d];
        float Bm = xd[row * XDW + DTR + n];
        float Cm = xd[row * XDW + DTR + NSTATE + n];
        h = expf(dt * A) * h + dt * Bm * x;
        float part = h * Cm;
#pragma unroll
        for (int off = 1; off < 16; off <<= 1)      // reduce within 16-lane half
            part += __shfl_xor(part, off, 32);
        if (n == 0) {
            float z  = xz[row * (2 * DI) + DI + d];
            float sz = z / (1.f + expf(-z));
            float y  = (part + Dr * x) * sz;
            dlt[row * DI + d] = y;
            yB[row * DI + d]  = (__bf16)y;
        }
    }
}

// ---------------------------------------------------------------- LN stats (one wave/row)
__global__ __launch_bounds__(256)
void k_lnstats(const float* __restrict__ tok, float* __restrict__ mu, float* __restrict__ rs) {
    int row  = blockIdx.x * (blockDim.x >> 5) + (threadIdx.x >> 5);
    int lane = threadIdx.x & 31;
    if (row >= MROWS) return;
    const float* t = tok + (size_t)row * D_MODEL;
    float s = 0.f, s2 = 0.f;
    for (int i = lane; i < D_MODEL; i += 32) { float v = t[i]; s += v; s2 += v * v; }
#pragma unroll
    for (int off = 1; off < 32; off <<= 1) { s += __shfl_xor(s, off, 32); s2 += __shfl_xor(s2, off, 32); }
    if (lane == 0) {
        float m = s / D_MODEL;
        mu[row] = m;
        rs[row] = rsqrtf(s2 / D_MODEL - m * m + 1e-5f);
    }
}

// ---------------------------------------------------------------- LN apply + mean over L
__global__ void k_final(const float* __restrict__ tok, const float* __restrict__ mu,
                        const float* __restrict__ rs, const float* __restrict__ g,
                        const float* __restrict__ bia, float* __restrict__ out) {
    int i = blockIdx.x * blockDim.x + threadIdx.x;
    if (i >= B_ * D_MODEL) return;
    int d = i % D_MODEL, b = i / D_MODEL;
    float acc = 0.f;
    for (int l = 0; l < LSEQ; ++l) {
        int row = b * LSEQ + l;
        acc += (tok[(size_t)row * D_MODEL + d] - mu[row]) * rs[row];
    }
    out[i] = g[d] * acc / (float)LSEQ + bia[d];
}

// ---------------------------------------------------------------- GEMM dispatch
static inline void launch_gemm(const __bf16* A, int lda, const __bf16* W, int ldw,
                               float* C, int ldc, __bf16* Cb,
                               int M, int N, int K, int addC, hipStream_t stream) {
    dim3 grid((N + BN - 1) / BN, (M + BM - 1) / BM);
    dim3 blk(256);
    const bool aligned = (M % BM == 0) && (N % BN == 0) && (K % BK == 0);
    if (aligned)
        k_gemm_bf16<false><<<grid, blk, 0, stream>>>(A, lda, W, ldw, C, ldc, Cb, M, N, K, addC);
    else
        k_gemm_bf16<true ><<<grid, blk, 0, stream>>>(A, lda, W, ldw, C, ldc, Cb, M, N, K, addC);
}

// ================================================================ launcher
extern "C" void kernel_launch(void* const* d_in, const int* in_sizes, int n_in,
                              void* d_out, int out_size, void* d_ws, size_t ws_size,
                              hipStream_t stream) {
    const float* x      = (const float*)d_in[0];
    const float* patchw = (const float*)d_in[1];
    const float* inw    = (const float*)d_in[2];
    const float* convw  = (const float*)d_in[3];
    const float* convb  = (const float*)d_in[4];
    const float* xprojw = (const float*)d_in[5];
    const float* dtw    = (const float*)d_in[6];
    const float* dtb    = (const float*)d_in[7];
    const float* Alog   = (const float*)d_in[8];
    const float* Dres   = (const float*)d_in[9];
    const float* outw   = (const float*)d_in[10];
    const float* lng    = (const float*)d_in[11];
    const float* lnb    = (const float*)d_in[12];
    float* out = (float*)d_out;

    // workspace bump allocator
    char*  ws  = (char*)d_ws;
    size_t off = 0;
    auto alloc = [&](size_t bytes) -> void* {
        void* p = ws + off;
        off = (off + bytes + 255) & ~(size_t)255;
        return p;
    };
    float*  tokF   = (float*) alloc((size_t)MROWS * D_MODEL * 4);
    __bf16* tokB   = (__bf16*)alloc((size_t)MROWS * D_MODEL * 2);
    __bf16* inwB   = (__bf16*)alloc((size_t)LAYERS * 2 * DI * D_MODEL * 2);
    __bf16* xprojB = (__bf16*)alloc((size_t)LAYERS * XDW * DI * 2);
    __bf16* dtwB   = (__bf16*)alloc((size_t)LAYERS * DI * DTR * 2);
    __bf16* outwB  = (__bf16*)alloc((size_t)LAYERS * D_MODEL * DI * 2);
    float*  xzF    = (float*) alloc((size_t)MROWS * 2 * DI * 4);
    float*  xrF    = (float*) alloc((size_t)MROWS * DI * 4);
    __bf16* xrB    = (__bf16*)alloc((size_t)MROWS * DI * 2);
    float*  xdF    = (float*) alloc((size_t)MROWS * XDW * 4);
    __bf16* xdB    = (__bf16*)alloc((size_t)MROWS * XDW * 2);
    float*  dltF   = (float*) alloc((size_t)MROWS * DI * 4);   // delta, then y
    __bf16* yB     = (__bf16*)alloc((size_t)MROWS * DI * 2);
    float*  muV    = (float*) alloc((size_t)MROWS * 4);
    float*  rsV    = (float*) alloc((size_t)MROWS * 4);

    const dim3 blk(256);
    auto grid1 = [](size_t n) { return dim3((unsigned)((n + 255) / 256)); };

    // weights -> bf16 (every call: deterministic, no caching)
    k_f32_to_bf16<<<grid1((size_t)LAYERS * 2 * DI * D_MODEL), blk, 0, stream>>>(inw, inwB, LAYERS * 2 * DI * D_MODEL);
    k_f32_to_bf16<<<grid1((size_t)LAYERS * XDW * DI), blk, 0, stream>>>(xprojw, xprojB, LAYERS * XDW * DI);
    k_f32_to_bf16<<<grid1((size_t)LAYERS * DI * DTR), blk, 0, stream>>>(dtw, dtwB, LAYERS * DI * DTR);
    k_f32_to_bf16<<<grid1((size_t)LAYERS * D_MODEL * DI), blk, 0, stream>>>(outw, outwB, LAYERS * D_MODEL * DI);

    // patchify
    k_patchify<<<grid1((size_t)MROWS * D_MODEL), blk, 0, stream>>>(x, patchw, tokF, tokB);

    const dim3 gScan((B_ * DI / 2 + 7) / 8);                               // 8 waves/block

    for (int L = 0; L < LAYERS; ++L) {
        const __bf16* inwL   = inwB   + (size_t)L * 2 * DI * D_MODEL;
        const __bf16* xprojL = xprojB + (size_t)L * XDW * DI;
        const __bf16* dtwL   = dtwB   + (size_t)L * DI * DTR;
        const __bf16* outwL  = outwB  + (size_t)L * D_MODEL * DI;
        const float*  convwL = convw  + (size_t)L * DI * KCONV;
        const float*  convbL = convb  + (size_t)L * DI;
        const float*  dtbL   = dtb    + (size_t)L * DI;
        const float*  AlogL  = Alog   + (size_t)L * DI * NSTATE;
        const float*  DresL  = Dres   + (size_t)L * DI;

        // xz = tok @ Wi^T                      (2048x768 x 768x3072)  [aligned]
        launch_gemm(tokB, D_MODEL, inwL, D_MODEL, xzF, 2 * DI, nullptr,
                    MROWS, 2 * DI, D_MODEL, 0, stream);
        // causal depthwise conv + SiLU -> xr
        k_convsilu<<<grid1((size_t)MROWS * DI), blk, 0, stream>>>(xzF, convwL, convbL, xrF, xrB);
        // x_dbl = xr @ Wx^T                    (2048x1536 x 1536x80)  [edge: N=80]
        launch_gemm(xrB, DI, xprojL, DI, xdF, XDW, xdB,
                    MROWS, XDW, DI, 0, stream);
        // delta_raw = dt @ Wdt^T               (2048x48 x 48x1536)    [edge: K=48]
        launch_gemm(xdB, XDW, dtwL, DTR, dltF, DI, nullptr,
                    MROWS, DI, DTR, 0, stream);
        // delta = softplus(delta_raw + dt_b)
        k_softplus<<<grid1((size_t)MROWS * DI), blk, 0, stream>>>(dltF, dtbL);
        // selective scan + D-residual + SiLU(z) gate -> y (in dltF) + bf16
        k_scan<<<gScan, blk, 0, stream>>>(dltF, xdF, xrF, xzF, AlogL, DresL, yB);
        // tok += y @ Wo^T                      (2048x1536 x 1536x768) [aligned]
        launch_gemm(yB, DI, outwL, DI, tokF, D_MODEL, tokB,
                    MROWS, D_MODEL, DI, 1, stream);
    }

    // final LayerNorm + mean over sequence
    k_lnstats<<<dim3((MROWS + 7) / 8), blk, 0, stream>>>(tokF, muV, rsV);
    k_final<<<grid1((size_t)B_ * D_MODEL), blk, 0, stream>>>(tokF, muV, rsV, lng, lnb, out);
    (void)in_sizes; (void)n_in; (void)out_size; (void)ws_size;
}
